// QuantumAttentionWithSimplifiedEntanglement_13709535609622
// MI455X (gfx1250) — matrix operand order repaired and checked by reading, not compiled, not run on hardware
//
#include <hip/hip_runtime.h>
#include <hip/hip_bf16.h>

// ---------------------------------------------------------------------------
// Quantum attention for MI455X (gfx1250, wave32, WMMA).
//   B=4, S=2048, F=256, N_QUBITS=4, N_LAYERS=3, ns=64 splits.
// Phase 1: fold each circuit's trainable part into a fixed complex 16x16
//          unitary per (set,split); embedding reduces to a real 16-vector r
//          with phase (-i)^popcount folded into the unitary.
// Phase 2: attention with v_wmma_f32_16x16x32_f16 (QK^T) and
//          v_wmma_f32_16x16x4_f32 (P*V).
// ---------------------------------------------------------------------------

typedef __attribute__((ext_vector_type(16))) _Float16 v16h;
typedef __attribute__((ext_vector_type(8)))  _Float16 v8h;
typedef __attribute__((ext_vector_type(8)))  float    v8f;
typedef __attribute__((ext_vector_type(2)))  float    v2f;

#define NB    4      // batches
#define SEQ   2048   // sequence length
#define FEAT  256    // features
#define NSPL  64     // splits
#define NSAMP 8192   // NB*SEQ

// ---------------------------------------------------------------------------
// Kernel 1: build the fixed 16x16 unitary per (set, split), fold the
// (-i)^popcount embedding phase into it, store Re/Im parts.
// grid = (64 splits, 3 sets), block = 16 threads (one per basis column).
// ---------------------------------------------------------------------------
__global__ void build_unitaries(const float* __restrict__ Wq, const float* __restrict__ Eq,
                                const float* __restrict__ Wk, const float* __restrict__ Ek,
                                const float* __restrict__ Wv, const float* __restrict__ Ev,
                                float* __restrict__ Vre, float* __restrict__ Vim) {
  const int split = blockIdx.x;
  const int set   = blockIdx.y;
  const float* W = (set == 0) ? Wq : (set == 1) ? Wk : Wv;
  const float* E = (set == 0) ? Eq : (set == 1) ? Ek : Ev;

  __shared__ float sre[16][16];   // [column][amplitude]
  __shared__ float sim[16][16];
  const int col = threadIdx.x;    // basis state this thread evolves

  for (int a = 0; a < 16; ++a) { sre[col][a] = (a == col) ? 1.f : 0.f; sim[col][a] = 0.f; }

  for (int l = 0; l < 3; ++l) {
    // rotation layer: RX(w) then RY(w) on each wire (same angle)
    for (int i = 0; i < 4; ++i) {
      float th = W[(split * 3 + l) * 4 + i];
      float s, c; sincosf(0.5f * th, &s, &c);
      const int m = 8 >> i;                 // wire i -> bit (3-i)
      for (int a = 0; a < 16; ++a) {        // RX: [[c,-is],[-is,c]]
        if (a & m) continue;
        const int a1 = a + m;
        float r0 = sre[col][a], i0 = sim[col][a];
        float r1 = sre[col][a1], i1 = sim[col][a1];
        sre[col][a]  = c * r0 + s * i1;  sim[col][a]  = c * i0 - s * r1;
        sre[col][a1] = c * r1 + s * i0;  sim[col][a1] = c * i1 - s * r0;
      }
      for (int a = 0; a < 16; ++a) {        // RY: [[c,-s],[s,c]]
        if (a & m) continue;
        const int a1 = a + m;
        float r0 = sre[col][a], i0 = sim[col][a];
        float r1 = sre[col][a1], i1 = sim[col][a1];
        sre[col][a]  = c * r0 - s * r1;  sim[col][a]  = c * i0 - s * i1;
        sre[col][a1] = s * r0 + c * r1;  sim[col][a1] = s * i0 + c * i1;
      }
    }
    // ring CRX(c=i, t=(i+1)%4)
    for (int i = 0; i < 4; ++i) {
      float th = E[(split * 3 + l) * 4 + i];
      float s, c; sincosf(0.5f * th, &s, &c);
      const int cm = 8 >> i;
      const int tm = 8 >> ((i + 1) & 3);
      for (int a = 0; a < 16; ++a) {
        if (!(a & cm)) continue;            // control bit must be 1
        if (a & tm) continue;               // target bit 0 half of the pair
        const int a1 = a + tm;
        float r0 = sre[col][a], i0 = sim[col][a];
        float r1 = sre[col][a1], i1 = sim[col][a1];
        sre[col][a]  = c * r0 + s * i1;  sim[col][a]  = c * i0 - s * r1;
        sre[col][a1] = c * r1 + s * i0;  sim[col][a1] = c * i1 - s * r0;
      }
    }
  }

  // fold (-i)^popcount(col) into this column, write V
  const int pc = __popc(col) & 3;
  for (int j = 0; j < 16; ++j) {
    float re = sre[col][j], im = sim[col][j], nr, ni;
    if      (pc == 0) { nr =  re; ni =  im; }
    else if (pc == 1) { nr =  im; ni = -re; }
    else if (pc == 2) { nr = -re; ni = -im; }
    else              { nr = -im; ni =  re; }
    Vre[((set * NSPL + split) * 16 + j) * 16 + col] = nr;
    Vim[((set * NSPL + split) * 16 + j) * 16 + col] = ni;
  }
}

// ---------------------------------------------------------------------------
// Kernel 2: per (sample, split) evaluate the three circuits.
// grid = (64 splits, 32 sample-chunks), block = 256 threads.
// Writes q,k row-major f16 [m][f]; v transposed f32 [batch][f][s].
// ---------------------------------------------------------------------------
__global__ void eval_circuits(const float* __restrict__ x,
                              const float* __restrict__ Vre, const float* __restrict__ Vim,
                              _Float16* __restrict__ qb, _Float16* __restrict__ kb,
                              float* __restrict__ vT) {
  const int split = blockIdx.x;
  const int tid   = threadIdx.x;
  __shared__ float Lre[3][256];
  __shared__ float Lim[3][256];
  for (int t = tid; t < 768; t += 256) {
    const int set = t >> 8, r = t & 255;
    Lre[set][r] = Vre[(set * NSPL + split) * 256 + r];
    Lim[set][r] = Vim[(set * NSPL + split) * 256 + r];
  }
  __syncthreads();

  const int m = blockIdx.y * 256 + tid;   // sample index 0..8191
  float cv[4], sv[4];
#pragma unroll
  for (int i = 0; i < 4; ++i) {
    const float t = x[(size_t)m * FEAT + split * 4 + i];
    sincosf(0.5f * t, &sv[i], &cv[i]);
  }
  float r[16];
#pragma unroll
  for (int b = 0; b < 16; ++b) {
    r[b] = ((b & 8) ? sv[0] : cv[0]) * ((b & 4) ? sv[1] : cv[1]) *
           ((b & 2) ? sv[2] : cv[2]) * ((b & 1) ? sv[3] : cv[3]);
  }

  for (int set = 0; set < 3; ++set) {
    float z0 = 0.f, z1 = 0.f, z2 = 0.f, z3 = 0.f;
#pragma unroll
    for (int j = 0; j < 16; ++j) {
      float re = 0.f, im = 0.f;
#pragma unroll
      for (int b = 0; b < 16; ++b) {
        re = fmaf(Lre[set][j * 16 + b], r[b], re);
        im = fmaf(Lim[set][j * 16 + b], r[b], im);
      }
      const float p = re * re + im * im;
      z0 += (j & 8) ? -p : p;
      z1 += (j & 4) ? -p : p;
      z2 += (j & 2) ? -p : p;
      z3 += (j & 1) ? -p : p;
    }
    if (set == 0) {
      _Float16* q = qb + (size_t)m * FEAT + split * 4;
      q[0] = (_Float16)z0; q[1] = (_Float16)z1; q[2] = (_Float16)z2; q[3] = (_Float16)z3;
    } else if (set == 1) {
      _Float16* k = kb + (size_t)m * FEAT + split * 4;
      k[0] = (_Float16)z0; k[1] = (_Float16)z1; k[2] = (_Float16)z2; k[3] = (_Float16)z3;
    } else {
      const int bb = m >> 11, ss = m & 2047;
      float* v = vT + ((size_t)bb * FEAT + split * 4) * SEQ + ss;
      v[0] = z0; v[SEQ] = z1; v[2 * SEQ] = z2; v[3 * SEQ] = z3;
    }
  }
}

// ---------------------------------------------------------------------------
// Kernel 3: raw scores = q k^T, one 16x16 tile per wave, f16 WMMA, K=256.
// 65536 tiles total; block = 128 threads (4 waves), grid = 16384.
// ---------------------------------------------------------------------------
union V16u { v16h v; v8h h[2]; };

__global__ void gemm_scores(const _Float16* __restrict__ qb,
                            const _Float16* __restrict__ kb,
                            float* __restrict__ attn) {
  const int wave = blockIdx.x * 4 + (threadIdx.x >> 5);
  const int b    = wave >> 14;            // 128*128 tiles per batch
  const int rem  = wave & 16383;
  const int tm   = rem >> 7;
  const int tn   = rem & 127;
  const int lane = threadIdx.x & 31;
  const int m16  = lane & 15;
  const int h    = lane >> 4;

  const _Float16* qrow = qb + (size_t)(b * SEQ + tm * 16 + m16) * FEAT;
  const _Float16* krow = kb + (size_t)(b * SEQ + tn * 16 + m16) * FEAT;

  v8f acc = {};
#pragma unroll
  for (int k0 = 0; k0 < FEAT; k0 += 32) {
    V16u a;
    a.v = *(const v16h*)(qrow + k0 + h * 16);         // A: K = h*16..+15
    const v16h bf = *(const v16h*)(krow + k0 + h * 16); // B: K = h*16..+15, col = m16
    acc = __builtin_amdgcn_wmma_f32_16x16x32_f16(false, a.v, false, bf,
                                                 (short)0, acc, false, false);
  }
  // D: lane(n=m16, h) holds rows r+8h
  float* dst = attn + (size_t)(b * SEQ + tm * 16 + 8 * h) * SEQ + tn * 16 + m16;
#pragma unroll
  for (int r = 0; r < 8; ++r) dst[(size_t)r * SEQ] = acc[r];
}

// ---------------------------------------------------------------------------
// Kernel 4: in-place row softmax of (raw_score * 0.5), row length 2048.
// grid = 8192 rows, block = 256 threads (8 elems/thread).
// ---------------------------------------------------------------------------
__global__ void softmax_rows(float* __restrict__ attn) {
  float* rp = attn + (size_t)blockIdx.x * SEQ;
  const int tid = threadIdx.x;
  __shared__ float red[256];

  float v[8], mx = -3.402823466e+38f;
#pragma unroll
  for (int j = 0; j < 8; ++j) { v[j] = rp[tid + j * 256] * 0.5f; mx = fmaxf(mx, v[j]); }
  red[tid] = mx; __syncthreads();
  for (int s = 128; s > 0; s >>= 1) { if (tid < s) red[tid] = fmaxf(red[tid], red[tid + s]); __syncthreads(); }
  mx = red[0]; __syncthreads();

  float sum = 0.f;
#pragma unroll
  for (int j = 0; j < 8; ++j) { v[j] = expf(v[j] - mx); sum += v[j]; }
  red[tid] = sum; __syncthreads();
  for (int s = 128; s > 0; s >>= 1) { if (tid < s) red[tid] += red[tid + s]; __syncthreads(); }
  const float inv = 1.f / red[0];
#pragma unroll
  for (int j = 0; j < 8; ++j) rp[tid + j * 256] = v[j] * inv;
}

// ---------------------------------------------------------------------------
// Kernel 5: out = attn @ v, full-precision f32 WMMA 16x16x4, K=2048.
// A = attn f32 row-major, B = vT f32 (so B loads are contiguous float2).
// 8192 tiles; block = 128 (4 waves), grid = 2048.
// ---------------------------------------------------------------------------
__global__ void gemm_out(const float* __restrict__ attn,
                         const float* __restrict__ vT,
                         float* __restrict__ out) {
  const int wave = blockIdx.x * 4 + (threadIdx.x >> 5);
  const int b    = wave >> 11;            // 128*16 tiles per batch
  const int rem  = wave & 2047;
  const int tm   = rem >> 4;
  const int tn   = rem & 15;
  const int lane = threadIdx.x & 31;
  const int m16  = lane & 15;
  const int h    = lane >> 4;

  const float* arow = attn + (size_t)(b * SEQ + tm * 16 + m16) * SEQ;
  const float* brow = vT   + ((size_t)b * FEAT + tn * 16 + m16) * SEQ;

  v8f acc = {};
#pragma unroll 8
  for (int k0 = 0; k0 < SEQ; k0 += 4) {
    const v2f a  = *(const v2f*)(arow + k0 + 2 * h);   // A[m][k0+2h], [k0+2h+1]
    const v2f bb = *(const v2f*)(brow + k0 + 2 * h);   // B[k0+2h][n], [k0+2h+1][n]
    acc = __builtin_amdgcn_wmma_f32_16x16x4_f32(false, a, false, bb,
                                                (short)0, acc, false, false);
  }
  float* dst = out + (size_t)(b * SEQ + tm * 16 + 8 * h) * FEAT + tn * 16 + m16;
#pragma unroll
  for (int r = 0; r < 8; ++r) dst[(size_t)r * FEAT] = acc[r];
}

// ---------------------------------------------------------------------------
extern "C" void kernel_launch(void* const* d_in, const int* in_sizes, int n_in,
                              void* d_out, int out_size, void* d_ws, size_t ws_size,
                              hipStream_t stream) {
  const float* x  = (const float*)d_in[0];
  const float* Wq = (const float*)d_in[1];
  const float* Eq = (const float*)d_in[2];
  const float* Wk = (const float*)d_in[3];
  const float* Ek = (const float*)d_in[4];
  const float* Wv = (const float*)d_in[5];
  const float* Ev = (const float*)d_in[6];

  float* out  = (float*)d_out;                        // (B,S,F)
  float* attn = out + (size_t)NB * SEQ * FEAT;        // (B,S,S)

  // workspace layout (all offsets 32B-aligned)
  char* ws = (char*)d_ws;
  float*    Vre = (float*)ws;                               //  49152 f32
  float*    Vim = Vre + 3 * NSPL * 256;                     //  49152 f32
  _Float16* qb  = (_Float16*)(ws + 2u * 3 * NSPL * 256 * 4);//  8192*256 f16
  _Float16* kb  = qb + (size_t)NSAMP * FEAT;                //  8192*256 f16
  float*    vT  = (float*)((char*)kb + (size_t)NSAMP * FEAT * 2); // 8192*256 f32 (transposed per batch)

  build_unitaries<<<dim3(NSPL, 3), 16, 0, stream>>>(Wq, Eq, Wk, Ek, Wv, Ev, Vre, Vim);
  eval_circuits<<<dim3(NSPL, NSAMP / 256), 256, 0, stream>>>(x, Vre, Vim, qb, kb, vT);
  gemm_scores<<<(NB * 128 * 128) / 4, 128, 0, stream>>>(qb, kb, attn);
  softmax_rows<<<NB * SEQ, 256, 0, stream>>>(attn);
  gemm_out<<<(NB * 128 * 16) / 4, 128, 0, stream>>>(attn, vT, out);
}
